// GATLayer_80839874445651
// MI455X (gfx1250) — matrix operand order repaired and checked by reading, not compiled, not run on hardware
//
#include <hip/hip_runtime.h>
#include <hip/hip_bf16.h>

// ---------------------------------------------------------------------------
// GAT layer, fused flash-style for MI455X (gfx1250, wave32, WMMA + TDM).
//   N=8192 nodes, IN_F=256, OUT_F=128, leaky slope 0.2, adj density 0.1.
// adj (268MB) is streamed exactly once (roofline floor ~11.5us @ 23.3TB/s);
// h (2MB bf16) stays in L2 (192MB). attention@h uses
// v_wmma_f32_16x16x32_bf16 with f32 accumulation. adj tiles are DMA'd by the
// Tensor Data Mover into LDS, double-buffered on TENSORcnt; last iteration is
// peeled so the hot loop has no branches / EXEC masking around the TDM issue.
// Softmax uses per-row constant m_i = leaky(s1_i + max_j s2_j) (upper bound),
// mathematically identical to the reference softmax and needing no online
// rescaling, so column-split partial sums combine by addition.
// ---------------------------------------------------------------------------

#define GAT_N     8192
#define GAT_INF   256
#define GAT_OUTF  128
#define GAT_ALPHA 0.2f

typedef __attribute__((ext_vector_type(16))) __bf16 v16bf;
typedef __attribute__((ext_vector_type(8)))  float  v8f;
typedef __attribute__((ext_vector_type(4)))  unsigned int v4u;
typedef __attribute__((ext_vector_type(8)))  int    v8i;
typedef __attribute__((ext_vector_type(4)))  int    v4i;

#if defined(__has_builtin)
#  if __has_builtin(__builtin_amdgcn_tensor_load_to_lds)
#    define GAT_USE_TDM 1
#  endif
#endif
#ifndef GAT_USE_TDM
#  define GAT_USE_TDM 0
#endif

// Padded LDS row stride for the 16x32 f32 adj tile: 32 data DWORDs + 4 pad
// DWORDs => 36. (16 lanes * 36 mod 64 banks covers all residues == 0 mod 4:
// conflict-free b128 reads per half-wave.)
#define TILE_ROW_DW 36
#define TILE_DW     (16 * TILE_ROW_DW)   // 576 DWORDs = 2304 B per buffer

#if GAT_USE_TDM
// Issue one TDM load: 2D tile 32(x) x 16(y) of 4-byte elements, row stride
// 8192 elements, LDS padding of 4 DWORDs every 32 DWORDs (ISA 8.3/8.4).
__device__ __forceinline__ void tdm_load_adj_tile(unsigned lds_addr,
                                                  unsigned long long gaddr)
{
    v4u g0;
    g0.x = 1u;                                   // count=1, user descriptor
    g0.y = lds_addr;                             // LDS byte address
    g0.z = (unsigned)(gaddr & 0xFFFFFFFFull);    // global_addr[31:0]
    g0.w = (unsigned)(gaddr >> 32) | (2u << 30); // global_addr[56:32] | type=2

    v8i g1;
    g1[0] = (int)((2u << 16)      // data_size = 4B
                | (1u << 20)      // pad_enable
                | (4u << 22)      // pad_interval: every 32 DWORDs
                | (3u << 25));    // pad_amount: 4 DWORDs
    g1[1] = (int)0x20000000u;     // tensor_dim0 = 8192 (low 16 in bits 31:16)
    g1[2] = (int)0x20000000u;     // dim0 hi = 0 | tensor_dim1 = 8192 (low 16)
    g1[3] = (int)(32u << 16);     // dim1 hi = 0 | tile_dim0 = 32
    g1[4] = 16;                   // tile_dim1 = 16, tile_dim2 = 0
    g1[5] = 8192;                 // tensor_dim0_stride = 8192 elements
    g1[6] = 0;                    // stride hi | tensor_dim1_stride lo
    g1[7] = 0;

    v4i gz = {0, 0, 0, 0};
#if defined(__clang_major__) && __clang_major__ >= 23
    v8i gz8 = {0, 0, 0, 0, 0, 0, 0, 0};
    __builtin_amdgcn_tensor_load_to_lds(g0, g1, gz, gz, gz8, 0);
#else
    __builtin_amdgcn_tensor_load_to_lds(g0, g1, gz, gz, 0);
#endif
}
#endif

// ---------------------------------------------------------------------------
// Kernel A: h = x @ W  (f32, exact), fused s1 = h@a1, s2 = h@a2,
// and write hT in bf16, transposed ([OUT_F][N]) so kernel B's WMMA B-tiles
// (K along rows of h) are contiguous 32-byte runs.
// ---------------------------------------------------------------------------
__global__ __launch_bounds__(256)
void gat_hproj(const float* __restrict__ x, const float* __restrict__ W,
               const float* __restrict__ a, unsigned short* __restrict__ hTbits,
               float* __restrict__ s1, float* __restrict__ s2)
{
    __shared__ float xT[256 * 36];      // xT[k][r], stride 36
    __shared__ float htile[32 * 132];   // htile[r][c], stride 132
    __shared__ float s1l[32], s2l[32];

    const int tid = threadIdx.x;
    const int i0  = blockIdx.x * 32;

    if (tid < 32) { s1l[tid] = 0.f; s2l[tid] = 0.f; }

    for (int idx = tid; idx < 32 * 256; idx += 256) {
        int r = idx >> 8;
        int k = idx & 255;
        xT[k * 36 + r] = x[(size_t)(i0 + r) * GAT_INF + k];
    }
    __syncthreads();

    const int tx = tid & 31;
    const int ty = tid >> 5;
    const int c0 = tx * 4;
    const int r0 = ty * 4;

    float acc[4][4];
#pragma unroll
    for (int r = 0; r < 4; ++r)
#pragma unroll
        for (int c = 0; c < 4; ++c) acc[r][c] = 0.f;

    for (int k = 0; k < GAT_INF; ++k) {
        float4 xv = *(const float4*)&xT[k * 36 + r0];
        float4 wv = *(const float4*)(W + (size_t)k * GAT_OUTF + c0);
        float xr[4] = { xv.x, xv.y, xv.z, xv.w };
        float wc[4] = { wv.x, wv.y, wv.z, wv.w };
#pragma unroll
        for (int r = 0; r < 4; ++r)
#pragma unroll
            for (int c = 0; c < 4; ++c) acc[r][c] = fmaf(xr[r], wc[c], acc[r][c]);
    }

    float a1c[4], a2c[4];
#pragma unroll
    for (int c = 0; c < 4; ++c) { a1c[c] = a[c0 + c]; a2c[c] = a[GAT_OUTF + c0 + c]; }
#pragma unroll
    for (int r = 0; r < 4; ++r) {
        float p1 = 0.f, p2 = 0.f;
#pragma unroll
        for (int c = 0; c < 4; ++c) {
            float hv = acc[r][c];
            htile[(r0 + r) * 132 + (c0 + c)] = hv;
            p1 = fmaf(hv, a1c[c], p1);
            p2 = fmaf(hv, a2c[c], p2);
        }
        atomicAdd(&s1l[r0 + r], p1);   // ds_add_f32
        atomicAdd(&s2l[r0 + r], p2);
    }
    __syncthreads();

    for (int idx = tid; idx < GAT_OUTF * 32; idx += 256) {
        int c  = idx >> 5;
        int rr = idx & 31;
        __bf16 b = (__bf16)htile[rr * 132 + c];
        unsigned short u;
        __builtin_memcpy(&u, &b, 2);
        hTbits[(size_t)c * GAT_N + i0 + rr] = u;
    }
    if (tid < 32) { s1[i0 + tid] = s1l[tid]; s2[i0 + tid] = s2l[tid]; }
}

// ---------------------------------------------------------------------------
// Kernel C: global max of s2 -> scalar
// ---------------------------------------------------------------------------
__global__ __launch_bounds__(256)
void gat_s2max(const float* __restrict__ s2, float* __restrict__ s2max)
{
    __shared__ float red[256];
    float m = -1e30f;
    for (int i = threadIdx.x; i < GAT_N; i += 256) m = fmaxf(m, s2[i]);
    red[threadIdx.x] = m;
    __syncthreads();
    for (int s = 128; s > 0; s >>= 1) {
        if (threadIdx.x < s) red[threadIdx.x] = fmaxf(red[threadIdx.x], red[threadIdx.x + s]);
        __syncthreads();
    }
    if (threadIdx.x == 0) *s2max = red[0];
}

// ---------------------------------------------------------------------------
// Kernel B: fused masked-softmax attention + attention@h via bf16 WMMA.
// Block = 4 waves = one 16-row tile; wave w covers columns [w*2048,(w+1)*2048).
// adj 16x32 tiles arrive via TDM (tensor_load_to_lds) into a per-wave double
// buffer gated by s_wait_tensorcnt; the last iteration is peeled so the
// steady-state loop is branch-free (no EXEC masking around the TDM issue).
// A-reg layout (16-bit A 16x32): lane m=l&15, half h=l>>4 holds
//   K = h*8+[0,8) and 16+h*8+[0,8).
// B-reg layout: lane n=l&15 holds K = h*8+[0,8) and 16+h*8+[0,8) from hT row n.
// ---------------------------------------------------------------------------
__global__ __launch_bounds__(128)
void gat_attn(const float* __restrict__ adj, const unsigned short* __restrict__ hTbits,
              const float* __restrict__ s1, const float* __restrict__ s2,
              const float* __restrict__ s2maxp, float* __restrict__ out)
{
    __shared__ float accs[4][8][32][8];   // [wave][nchunk][lane][reg] = 32 KB
    __shared__ float dens[4][32];
#if GAT_USE_TDM
    __shared__ float adjtile[4][2][TILE_DW];   // per-wave double buffer, 18 KB
#endif

    const int tid  = threadIdx.x;
    const int lane = tid & 31;
    const int l16  = lane & 15;
    const int hh   = lane >> 4;
    // wave index is wave-uniform by construction; make it provably scalar so
    // loop bounds and TDM descriptor addresses live in SGPRs.
    const int wave = __builtin_amdgcn_readfirstlane(tid >> 5);
    const int row0 = blockIdx.x * 16;
    const int myrow = row0 + l16;

    const __bf16* hT = (const __bf16*)hTbits;

    const float s1v  = s1[myrow];
    const float s2m  = *s2maxp;
    const float zmax = s1v + s2m;
    const float mi   = zmax > 0.f ? zmax : GAT_ALPHA * zmax;   // >= row max of e

    const float* adjrow = adj + (size_t)myrow * GAT_N;

    v8f acc[8];
    const v8f vzero = {0.f, 0.f, 0.f, 0.f, 0.f, 0.f, 0.f, 0.f};
#pragma unroll
    for (int nc = 0; nc < 8; ++nc) acc[nc] = vzero;

    float den0 = 0.f, den1 = 0.f, den2 = 0.f, den3 = 0.f;
    const int jbeg = wave * (GAT_N / 4);
    constexpr int NIT = (GAT_N / 4) / 32;   // 64 iterations of 32 columns

    // One 32-column step: adj tile (A operand) from LDS (TDM-filled) or
    // global, s2 slice from global (L0-hot), exp+pack, 8 WMMAs.
    auto body = [&](const float4& A0, const float4& A1,
                    const float4& A2, const float4& A3, int j) {
        const float* sp = s2 + j + hh * 8;
        float4 S0 = *(const float4*)(sp);
        float4 S1 = *(const float4*)(sp + 4);
        float4 S2 = *(const float4*)(sp + 16);
        float4 S3 = *(const float4*)(sp + 20);

        float av[16] = { A0.x, A0.y, A0.z, A0.w, A1.x, A1.y, A1.z, A1.w,
                         A2.x, A2.y, A2.z, A2.w, A3.x, A3.y, A3.z, A3.w };
        float sv[16] = { S0.x, S0.y, S0.z, S0.w, S1.x, S1.y, S1.z, S1.w,
                         S2.x, S2.y, S2.z, S2.w, S3.x, S3.y, S3.z, S3.w };

        v16bf Am;
        float pv[16];
#pragma unroll
        for (int e = 0; e < 16; ++e) {
            float z  = s1v + sv[e];
            float lr = z > 0.f ? z : GAT_ALPHA * z;
            float p  = av[e] > 0.f ? __expf(lr - mi) : 0.f;
            pv[e] = p;
            Am[e] = (__bf16)p;
        }
#pragma unroll
        for (int e = 0; e < 16; e += 4) {
            den0 += pv[e + 0]; den1 += pv[e + 1];
            den2 += pv[e + 2]; den3 += pv[e + 3];
        }

        const __bf16* hbase = hT + (size_t)(j + hh * 8);
#pragma unroll
        for (int nc = 0; nc < 8; ++nc) {
            const __bf16* bp = hbase + (size_t)(nc * 16 + l16) * GAT_N;
            uint4 B0 = *(const uint4*)(bp);
            uint4 B1 = *(const uint4*)(bp + 16);
            v16bf Bm;
            __builtin_memcpy(&Bm, &B0, 16);
            __builtin_memcpy((char*)&Bm + 16, &B1, 16);
            acc[nc] = __builtin_amdgcn_wmma_f32_16x16x32_bf16(
                false, Am, false, Bm, (short)0, acc[nc], false, false);
        }
    };

#if GAT_USE_TDM
    const unsigned lds0 = (unsigned)(unsigned long long)(uintptr_t)&adjtile[wave][0][0];
    const unsigned long long gbase =
        (unsigned long long)(uintptr_t)adj + ((unsigned long long)row0 * GAT_N) * 4ull;
    const float* tbase = &adjtile[wave][0][l16 * TILE_ROW_DW + hh * 8];

    // prologue: tile 0 in flight
    tdm_load_adj_tile(lds0, gbase + (unsigned long long)jbeg * 4ull);

    int j = jbeg;
    for (int itr = 0; itr < NIT - 1; ++itr, j += 32) {
        const int buf = itr & 1;
        // issue next tile, then wait until only it is outstanding
        tdm_load_adj_tile(lds0 + (unsigned)((buf ^ 1) * TILE_DW * 4),
                          gbase + (unsigned long long)(j + 32) * 4ull);
        __builtin_amdgcn_s_wait_tensorcnt((short)1);
        __asm__ volatile("" ::: "memory");    // tile reads stay below the wait
        __builtin_prefetch(adjrow + j + 2048, 0, 0);   // warm GL2 ahead of TDM

        const float* tp = tbase + buf * TILE_DW;
        float4 A0 = *(const float4*)(tp);
        float4 A1 = *(const float4*)(tp + 4);
        float4 A2 = *(const float4*)(tp + 16);
        float4 A3 = *(const float4*)(tp + 20);
        body(A0, A1, A2, A3, j);
    }
    // peeled last iteration: drain TENSORcnt, no further issue
    {
        __builtin_amdgcn_s_wait_tensorcnt((short)0);
        __asm__ volatile("" ::: "memory");
        const float* tp = tbase + ((NIT - 1) & 1) * TILE_DW;
        float4 A0 = *(const float4*)(tp);
        float4 A1 = *(const float4*)(tp + 4);
        float4 A2 = *(const float4*)(tp + 16);
        float4 A3 = *(const float4*)(tp + 20);
        body(A0, A1, A2, A3, j);
    }
#else
    int j = jbeg;
    for (int itr = 0; itr < NIT; ++itr, j += 32) {
        const float* ap = adjrow + j + hh * 8;
        __builtin_prefetch(ap + 512, 0, 0);
        float4 A0 = *(const float4*)(ap);
        float4 A1 = *(const float4*)(ap + 4);
        float4 A2 = *(const float4*)(ap + 16);
        float4 A3 = *(const float4*)(ap + 20);
        body(A0, A1, A2, A3, j);
    }
#endif

    float denom = (den0 + den1) + (den2 + den3);
    // lane l and l^16 hold complementary K-halves of row l&15's denominator
    denom += __shfl_xor(denom, 16, 32);
    dens[wave][lane] = denom;
#pragma unroll
    for (int nc = 0; nc < 8; ++nc)
#pragma unroll
        for (int r = 0; r < 8; ++r) accs[wave][nc][lane][r] = acc[nc][r];
    __syncthreads();

    // combine 4 wave-partials; wave w finalizes n-chunks {2w, 2w+1}
    float dr[8];
#pragma unroll
    for (int r = 0; r < 8; ++r) {
        int m = r + hh * 8;                          // output row for C/D reg r
        float d = dens[0][m] + dens[1][m] + dens[2][m] + dens[3][m];
        dr[r] = 1.f / fmaxf(d, 1e-30f);
    }
#pragma unroll
    for (int q = 0; q < 2; ++q) {
        int cc = wave * 2 + q;
#pragma unroll
        for (int r = 0; r < 8; ++r) {
            float v = accs[0][cc][lane][r] + accs[1][cc][lane][r] +
                      accs[2][cc][lane][r] + accs[3][cc][lane][r];
            int m = r + hh * 8;
            out[(size_t)(row0 + m) * GAT_OUTF + cc * 16 + l16] = v * dr[r];
        }
    }
}

// ---------------------------------------------------------------------------
// Launch: inputs (setup_inputs order): x, adj, W, a. Output: h_prime [N,128] f32.
// Workspace: hT bf16 (2MB) | s1 (32KB) | s2 (32KB) | s2max (4B).
// ---------------------------------------------------------------------------
extern "C" void kernel_launch(void* const* d_in, const int* in_sizes, int n_in,
                              void* d_out, int out_size, void* d_ws, size_t ws_size,
                              hipStream_t stream)
{
    (void)in_sizes; (void)n_in; (void)out_size; (void)ws_size;
    const float* x   = (const float*)d_in[0];
    const float* adj = (const float*)d_in[1];
    const float* W   = (const float*)d_in[2];
    const float* a   = (const float*)d_in[3];
    float* out = (float*)d_out;

    char* ws = (char*)d_ws;
    unsigned short* hT = (unsigned short*)ws;                         // 128*8192*2B
    float* s1  = (float*)(ws + (size_t)GAT_OUTF * GAT_N * 2);
    float* s2  = s1 + GAT_N;
    float* s2m = s2 + GAT_N;

    gat_hproj<<<GAT_N / 32, 256, 0, stream>>>(x, W, a, hT, s1, s2);
    gat_s2max<<<1, 256, 0, stream>>>(s2, s2m);
    gat_attn <<<GAT_N / 16, 128, 0, stream>>>(adj, hT, s1, s2, s2m, out);
}
